// Liear_MSA_me_29042568856120
// MI455X (gfx1250) — matrix operand order repaired and checked by reading, not compiled, not run on hardware
//
#include <hip/hip_runtime.h>
#include <hip/hip_bf16.h>

// ---------------------------------------------------------------------------
// Problem constants (from reference)
// ---------------------------------------------------------------------------
static constexpr int HEADS = 8, DIM = 8, T = 64;       // T = HEADS*DIM
static constexpr int C_IN = 256, C_OUT = 256;
static constexpr int B_ = 8, H_ = 128, W_ = 128;
static constexpr int HW = H_ * W_;                      // 16384
static constexpr int MROWS = 3 * T * 2;                 // 384 = [qkv(192); p(192)]
static constexpr int TILE_N = 64;
static constexpr int NTILES = (B_ * HW) / TILE_N;       // 2048

typedef _Float16 v16h __attribute__((ext_vector_type(16)));
typedef _Float16 v8h  __attribute__((ext_vector_type(8)));
typedef float    v8f  __attribute__((ext_vector_type(8)));

__device__ __forceinline__ v16h cat8(v8h lo, v8h hi) {
  return __builtin_shufflevector(lo, hi, 0,1,2,3,4,5,6,7,8,9,10,11,12,13,14,15);
}

// A-matrix fragment (16x32 f16, row M = lane&15).
// lanes 0-15: K = [0..7, 16..23]; lanes 16-31: K = [8..15, 24..31].
__device__ __forceinline__ v16h load_a_frag(const _Float16* rowk, int lane) {
  int kh = (lane & 16) ? 8 : 0;
  v8h lo = *(const v8h*)(rowk + kh);
  v8h hi = *(const v8h*)(rowk + kh + 16);
  return cat8(lo, hi);
}

// B-matrix fragment (32x16 f16, col N = lane&15), tile stored [n][k].
// lanes 0-15: K = [0..7, 16..23]; lanes 16-31: K = [8..15, 24..31].
__device__ __forceinline__ v16h load_b_frag(const _Float16* colk, int lane) {
  int kb = (lane & 16) ? 8 : 0;
  v8h lo = *(const v8h*)(colk + kb);
  v8h hi = *(const v8h*)(colk + kb + 16);
  return cat8(lo, hi);
}

__device__ __forceinline__ v8f wmma_f16(v16h a, v16h b, v8f c) {
  return __builtin_amdgcn_wmma_f32_16x16x32_f16(false, a, false, b, (short)0, c,
                                                false, false);
}

// Async DMA: copy 16B (and 16B at +16) from global to LDS; ASYNCcnt-tracked.
// ISA: the instruction offset is added to both the LDS and the global address.
__device__ __forceinline__ void async_copy32(unsigned lds_off, const void* g) {
  asm volatile("global_load_async_to_lds_b128 %0, %1, off"
               :: "v"(lds_off), "v"(g) : "memory");
  asm volatile("global_load_async_to_lds_b128 %0, %1, off offset:16"
               :: "v"(lds_off), "v"(g) : "memory");
}

__device__ __forceinline__ void wait_async0() {
  asm volatile("s_wait_asynccnt 0x0" ::: "memory");
}

// ---------------------------------------------------------------------------
// Prep: combined weight Wcat (384x256 f16): rows 0..191 = W_qkv,
// rows 192..383 = blockdiag(W_p) @ W_qkv  (folds the per-head projection).
// ---------------------------------------------------------------------------
__global__ void k_prep_wcat(const float* __restrict__ Wqkv,
                            const float* __restrict__ Wp,
                            _Float16* __restrict__ Wcat) {
  int idx = blockIdx.x * blockDim.x + threadIdx.x;
  if (idx >= MROWS * C_IN) return;
  int r = idx >> 8, c = idx & 255;
  float v;
  if (r < 192) {
    v = Wqkv[r * C_IN + c];
  } else {
    int r2 = r - 192, g = r2 >> 3, o = r2 & 7;
    float acc = 0.f;
#pragma unroll
    for (int i = 0; i < 8; ++i)
      acc += Wp[g * 64 + o * 8 + i] * Wqkv[(g * 8 + i) * C_IN + c];
    v = acc;
  }
  Wcat[idx] = (_Float16)v;
}

__global__ void k_prep_wffn(const float* __restrict__ Wffn_f32,
                            _Float16* __restrict__ Wffn) {
  int idx = blockIdx.x * blockDim.x + threadIdx.x;
  if (idx < C_OUT * 2 * T) Wffn[idx] = (_Float16)Wffn_f32[idx];
}

// ---------------------------------------------------------------------------
// K1: WMMA GEMM  gpre[b, 0:384, hw] = Wcat(384x256) @ x[b, :, hw]   (f16 out)
// Block: 256 threads (8 waves). Each block handles 384 x 64 tiles.
// x tile staged with global_load_async_to_lds_b128 (ASYNCcnt), then an
// LDS->LDS f32->f16 convert/transpose pass feeds the WMMA B-fragment layout.
// ---------------------------------------------------------------------------
__global__ __launch_bounds__(256) void k_gemm_qkvp(
    const float* __restrict__ x, const _Float16* __restrict__ Wcat,
    _Float16* __restrict__ gpre) {
  __shared__ __align__(16) float    sXf[32 * TILE_N];   // raw f32 tile, 8 KB
  __shared__ __align__(16) _Float16 sX[TILE_N * 32];    // [n][k] f16, 4 KB
  const int lane = threadIdx.x & 31;
  const int wave = threadIdx.x >> 5;
  const int lrow = lane & 15;

  // thread's fixed role in staging / convert passes
  const int srow = threadIdx.x >> 3;             // 0..31  (K row)
  const int sseg = (threadIdx.x & 7) << 3;       // 0..56  (8 floats)
  const unsigned lds_dst =
      (unsigned)(unsigned long long)(const void*)&sXf[srow * TILE_N + sseg];

  for (int tile = blockIdx.x; tile < NTILES; tile += gridDim.x) {
    const int b = tile >> 8;               // 256 tiles per batch image
    const int hw0 = (tile & 255) << 6;
    const float* xb = x + (size_t)b * C_IN * HW;

    v8f acc[12];
#pragma unroll
    for (int i = 0; i < 12; ++i) acc[i] = v8f{0.f,0.f,0.f,0.f,0.f,0.f,0.f,0.f};

    for (int k0 = 0; k0 < C_IN; k0 += 32) {
      __syncthreads();  // previous iteration's reads of sXf/sX are done
      {  // async DMA x[k0..k0+31][hw0..hw0+63] (f32, row-major) into sXf
        const float* src = xb + (size_t)(k0 + srow) * HW + hw0 + sseg;
        async_copy32(lds_dst, src);
        if (k0 + 32 < C_IN) __builtin_prefetch(src + 32 * HW, 0, 1);
        wait_async0();
      }
      __syncthreads();
      {  // LDS->LDS convert/transpose: sXf[k][n] (f32) -> sX[n][k] (f16)
        const float* sr = &sXf[srow * TILE_N + sseg];
        float4 f0 = *(const float4*)(sr);
        float4 f1 = *(const float4*)(sr + 4);
#pragma unroll
        for (int j = 0; j < 4; ++j) {
          sX[(sseg + j) * 32 + srow]     = (_Float16)(&f0.x)[j];
          sX[(sseg + 4 + j) * 32 + srow] = (_Float16)(&f1.x)[j];
        }
      }
      __syncthreads();

      v16h af[3], bf[4];
#pragma unroll
      for (int i = 0; i < 3; ++i) {
        int mrow = (wave * 3 + i) * 16 + lrow;
        af[i] = load_a_frag(Wcat + (size_t)mrow * C_IN + k0, lane);
      }
#pragma unroll
      for (int j = 0; j < 4; ++j)
        bf[j] = load_b_frag(&sX[(j * 16 + lrow) * 32], lane);
#pragma unroll
      for (int i = 0; i < 3; ++i)
#pragma unroll
        for (int j = 0; j < 4; ++j)
          acc[i * 4 + j] = wmma_f16(af[i], bf[j], acc[i * 4 + j]);
    }

    _Float16* gp = gpre + (size_t)b * MROWS * HW + hw0;
#pragma unroll
    for (int i = 0; i < 3; ++i)
#pragma unroll
      for (int j = 0; j < 4; ++j) {
        int n = j * 16 + (lane & 15);
#pragma unroll
        for (int r = 0; r < 8; ++r) {
          int m = (wave * 3 + i) * 16 + ((lane < 16) ? r : 8 + r);
          gp[(size_t)m * HW + n] = (_Float16)acc[i * 4 + j][r];
        }
      }
  }
}

// ---------------------------------------------------------------------------
// K2: depthwise 3x3 SAME conv on the p-rows (gpre rows 192..383) -> dbuf f16
// ---------------------------------------------------------------------------
__global__ __launch_bounds__(256) void k_dwconv(
    const _Float16* __restrict__ gpre, const float* __restrict__ Wd,
    _Float16* __restrict__ dbuf) {
  int idx = blockIdx.x * blockDim.x + threadIdx.x;
  const int total = B_ * 192 * H_ * (W_ / 8);
  if (idx >= total) return;
  int t = idx;
  int xs = (t & 15) << 3;  t >>= 4;
  int y  = t & 127;        t >>= 7;
  int c  = t % 192;
  int b  = t / 192;

  const _Float16* p = gpre + ((size_t)b * MROWS + 192 + c) * HW;
  float w[9];
#pragma unroll
  for (int i = 0; i < 9; ++i) w[i] = Wd[c * 9 + i];

  float out[8];
#pragma unroll
  for (int j = 0; j < 8; ++j) out[j] = 0.f;

#pragma unroll
  for (int dy = -1; dy <= 1; ++dy) {
    int yy = y + dy;
    if (yy < 0 || yy > H_ - 1) continue;
    const _Float16* row = p + yy * W_;
#pragma unroll
    for (int dx = -1; dx <= 1; ++dx) {
      float wv = w[(dy + 1) * 3 + (dx + 1)];
#pragma unroll
      for (int j = 0; j < 8; ++j) {
        int xx = xs + j + dx;
        if (xx >= 0 && xx < W_) out[j] += wv * (float)row[xx];
      }
    }
  }
  _Float16* dst = dbuf + ((size_t)b * 192 + c) * HW + y * W_ + xs;
#pragma unroll
  for (int j = 0; j < 8; ++j) dst[j] = (_Float16)out[j];
}

// ---------------------------------------------------------------------------
// K3: kv[b,g,e,d'] = sum_n relu(k)[n,e] * [v[n,d'] | 1]   (e in 0..7, d' in 0..8)
// One block per (b,g); block-local LDS reduction (LDS float atomicAdd, no g-atomics).
// ---------------------------------------------------------------------------
__global__ __launch_bounds__(256) void k_kv(const _Float16* __restrict__ gpre,
                                            const _Float16* __restrict__ dbuf,
                                            float* __restrict__ kv) {
  const int b = blockIdx.x >> 4;
  const int g = blockIdx.x & 15;
  const _Float16* base =
      (g < 8) ? gpre + ((size_t)b * MROWS + g * 24) * HW
              : dbuf + ((size_t)b * 192 + (g - 8) * 24) * HW;

  float acc[72];
#pragma unroll
  for (int i = 0; i < 72; ++i) acc[i] = 0.f;

  for (int it = 0; it < 8; ++it) {
    int hw = (it * 256 + threadIdx.x) << 3;  // 8 consecutive positions
    v8h ch[16];
#pragma unroll
    for (int s = 0; s < 16; ++s)
      ch[s] = *(const v8h*)(base + (size_t)(8 + s) * HW + hw);
#pragma unroll
    for (int p = 0; p < 8; ++p) {
      float kk[8], vv[8];
#pragma unroll
      for (int e = 0; e < 8; ++e) kk[e] = fmaxf((float)ch[e][p], 0.f);
#pragma unroll
      for (int d = 0; d < 8; ++d) vv[d] = (float)ch[8 + d][p];
#pragma unroll
      for (int e = 0; e < 8; ++e) {
#pragma unroll
        for (int d = 0; d < 8; ++d) acc[e * 9 + d] += kk[e] * vv[d];
        acc[e * 9 + 8] += kk[e];
      }
    }
  }

  __shared__ float skv[72];
  if (threadIdx.x < 72) skv[threadIdx.x] = 0.f;
  __syncthreads();
#pragma unroll
  for (int i = 0; i < 72; ++i) atomicAdd(&skv[i], acc[i]);
  __syncthreads();
  if (threadIdx.x < 72) kv[(size_t)blockIdx.x * 72 + threadIdx.x] = skv[threadIdx.x];
}

// ---------------------------------------------------------------------------
// K4: per 64-position tile: build o (128 ch, f16) in LDS from q·kv / denom,
// then WMMA GEMM y = W_ffn(256x128) @ o with fused batch-norm, f32 out.
// ---------------------------------------------------------------------------
__global__ __launch_bounds__(256) void k_attn_ffn(
    const _Float16* __restrict__ gpre, const _Float16* __restrict__ dbuf,
    const float* __restrict__ kv, const _Float16* __restrict__ Wffn,
    const float* __restrict__ bn_gamma, const float* __restrict__ bn_beta,
    const float* __restrict__ bn_mean, const float* __restrict__ bn_var,
    float* __restrict__ out) {
  __shared__ __align__(16) _Float16 sO[TILE_N * 128];  // [n][c2], 16 KB
  __shared__ float sKV[16 * 72];
  __shared__ float sScale[C_OUT], sBias[C_OUT];
  const int lane = threadIdx.x & 31;
  const int wave = threadIdx.x >> 5;
  const int lrow = lane & 15;

  for (int i = threadIdx.x; i < C_OUT; i += 256) {
    float sc = bn_gamma[i] * rsqrtf(bn_var[i] + 1e-5f);
    sScale[i] = sc;
    sBias[i] = bn_beta[i] - bn_mean[i] * sc;
  }

  int prev_b = -1;
  for (int tile = blockIdx.x; tile < NTILES; tile += gridDim.x) {
    const int b = tile >> 8;
    const int hw0 = (tile & 255) << 6;
    if (b != prev_b) {
      __syncthreads();
      for (int i = threadIdx.x; i < 16 * 72; i += 256)
        sKV[i] = kv[(size_t)b * 16 * 72 + i];
      prev_b = b;
    }
    __syncthreads();

    // --- build o tile: 64 positions x 16 groups, 4 tasks/thread -----------
    for (int task = threadIdx.x; task < 1024; task += 256) {
      int n = task & 63, g = task >> 6;
      const _Float16* base =
          (g < 8) ? gpre + ((size_t)b * MROWS + g * 24) * HW
                  : dbuf + ((size_t)b * 192 + (g - 8) * 24) * HW;
      int hw = hw0 + n;
      float q[8];
#pragma unroll
      for (int e = 0; e < 8; ++e)
        q[e] = fmaxf((float)base[(size_t)e * HW + hw], 0.f);
      const float* kvg = &sKV[g * 72];
      float den = 0.f;
#pragma unroll
      for (int e = 0; e < 8; ++e) den += q[e] * kvg[e * 9 + 8];
      float inv = 1.f / (den + 1e-15f);
#pragma unroll
      for (int d = 0; d < 8; ++d) {
        float a = 0.f;
#pragma unroll
        for (int e = 0; e < 8; ++e) a += q[e] * kvg[e * 9 + d];
        sO[n * 128 + g * 8 + d] = (_Float16)(a * inv);
      }
    }
    __syncthreads();

    // --- y = W_ffn @ o, 16 M-tiles x 4 N-tiles, 8 C-tiles per wave --------
    v8f acc[8];
#pragma unroll
    for (int i = 0; i < 8; ++i) acc[i] = v8f{0.f,0.f,0.f,0.f,0.f,0.f,0.f,0.f};

    for (int k0 = 0; k0 < 128; k0 += 32) {
      v16h af[2], bf[4];
#pragma unroll
      for (int i = 0; i < 2; ++i) {
        int mrow = (wave * 2 + i) * 16 + lrow;
        af[i] = load_a_frag(Wffn + (size_t)mrow * 128 + k0, lane);
      }
#pragma unroll
      for (int j = 0; j < 4; ++j)
        bf[j] = load_b_frag(&sO[(j * 16 + lrow) * 128 + k0], lane);
#pragma unroll
      for (int i = 0; i < 2; ++i)
#pragma unroll
        for (int j = 0; j < 4; ++j)
          acc[i * 4 + j] = wmma_f16(af[i], bf[j], acc[i * 4 + j]);
    }

    float* outb = out + (size_t)b * C_OUT * HW + hw0;
#pragma unroll
    for (int i = 0; i < 2; ++i)
#pragma unroll
      for (int j = 0; j < 4; ++j) {
        int n = j * 16 + (lane & 15);
#pragma unroll
        for (int r = 0; r < 8; ++r) {
          int m = (wave * 2 + i) * 16 + ((lane < 16) ? r : 8 + r);
          outb[(size_t)m * HW + n] = acc[i * 4 + j][r] * sScale[m] + sBias[m];
        }
      }
    __syncthreads();  // sO reused next tile
  }
}

// ---------------------------------------------------------------------------
// Launch
// ---------------------------------------------------------------------------
extern "C" void kernel_launch(void* const* d_in, const int* in_sizes, int n_in,
                              void* d_out, int out_size, void* d_ws,
                              size_t ws_size, hipStream_t stream) {
  const float* x        = (const float*)d_in[0];
  const float* W_qkv    = (const float*)d_in[1];
  const float* W_p      = (const float*)d_in[2];
  const float* W_d      = (const float*)d_in[3];
  const float* W_ffn    = (const float*)d_in[4];
  const float* bn_gamma = (const float*)d_in[5];
  const float* bn_beta  = (const float*)d_in[6];
  const float* bn_mean  = (const float*)d_in[7];
  const float* bn_var   = (const float*)d_in[8];
  float* out = (float*)d_out;

  char* ws = (char*)d_ws;
  _Float16* Wcat = (_Float16*)(ws);                      // 384*256*2   = 196608
  _Float16* Wffn = (_Float16*)(ws + 196608);             // 256*128*2   =  65536
  float*    kvb  = (float*)(ws + 262144);                // 8*16*72*4   =  36864
  _Float16* gpre = (_Float16*)(ws + 299008);             // 8*384*HW*2  = 100663296
  _Float16* dbuf = (_Float16*)(ws + 299008 + 100663296); // 8*192*HW*2  =  50331648

  k_prep_wcat<<<(MROWS * C_IN + 255) / 256, 256, 0, stream>>>(W_qkv, W_p, Wcat);
  k_prep_wffn<<<(C_OUT * 2 * T + 255) / 256, 256, 0, stream>>>(W_ffn, Wffn);
  k_gemm_qkvp<<<1024, 256, 0, stream>>>(x, Wcat, gpre);
  k_dwconv<<<(B_ * 192 * H_ * (W_ / 8) + 255) / 256, 256, 0, stream>>>(gpre, W_d, dbuf);
  k_kv<<<B_ * 16, 256, 0, stream>>>(gpre, dbuf, kvb);
  k_attn_ffn<<<1024, 256, 0, stream>>>(gpre, dbuf, kvb, Wffn, bn_gamma, bn_beta,
                                       bn_mean, bn_var, out);
}
